// MultiHeadSelfAttention_28398323761361
// MI455X (gfx1250) — matrix operand order repaired and checked by reading, not compiled, not run on hardware
//
#include <hip/hip_runtime.h>
#include <hip/hip_bf16.h>

// ---------------------------------------------------------------------------
// MHSA on gfx1250: bf16 WMMA GEMMs + flash-attention, fp32 accumulate.
// Async LDS staging via GLOBAL_LOAD_ASYNC_TO_LDS_B128 (ASYNCcnt).
// B=4, S=2048, D_MODEL=1024, H=16, DEPTH=64.
// ---------------------------------------------------------------------------

typedef __attribute__((ext_vector_type(16))) __bf16 v16bf;
typedef __attribute__((ext_vector_type(8)))  float  v8f;

#define BATCH   4
#define SEQ     2048
#define DMODEL  1024
#define NHEADS  16
#define DEPTH   64
#define MROWS   (BATCH * SEQ)   // 8192

__device__ __forceinline__ __bf16 f2bf(float x) {
    union { float f; unsigned u; } a; a.f = x;
    unsigned r = a.u + 0x7FFFu + ((a.u >> 16) & 1u);
    union { unsigned short s; __bf16 b; } o; o.s = (unsigned short)(r >> 16);
    return o.b;
}

__device__ __forceinline__ v8f vzero8() {
    v8f z;
#pragma unroll
    for (int i = 0; i < 8; ++i) z[i] = 0.0f;
    return z;
}

// Flat shared address -> 32-bit LDS byte offset (ISA 10.2: LDS aperture uses
// addr[31:0] directly).
__device__ __forceinline__ unsigned lds_off(const void* p) {
    return (unsigned)(unsigned long long)p;
}

// Per-lane async copy of 16 bytes: LDS[lds] = global[base + off .. +15].
// GVS addressing: vdst = LDS addr, vaddr = 32-bit byte offset, saddr = base.
__device__ __forceinline__ void async_b128(unsigned lds, unsigned off,
                                           unsigned long long base) {
    asm volatile("global_load_async_to_lds_b128 %0, %1, %2"
                 :: "v"(lds), "v"(off), "s"(base) : "memory");
}
__device__ __forceinline__ void wait_async0() {
    asm volatile("s_wait_asynccnt 0x0" ::: "memory");
}

// A-fragment (16x32, 16-bit): lane = M (lane&15), per-lane K chunks
// {0..7} and {16..23}, +8 for lanes 16-31.  (ISA 7.12.2)
__device__ __forceinline__ v16bf load_frag_a(const __bf16* base, int ld, int lane) {
    const __bf16* p = base + (lane & 15) * ld + ((lane >> 4) << 3);
    union { v16bf v; float4 q[2]; } u;
    u.q[0] = *(const float4*)(p);
    u.q[1] = *(const float4*)(p + 16);
    return u.v;
}

// B-fragment (32x16, 16-bit): lane = N (lane&15), per-lane K chunks
// {0..7} and {16..23}, +8 for lanes 16-31.  Source laid out [N][K].
__device__ __forceinline__ v16bf load_frag_b(const __bf16* base, int ld, int lane) {
    const __bf16* p = base + (lane & 15) * ld + ((lane >> 4) << 3);
    union { v16bf v; float4 q[2]; } u;
    u.q[0] = *(const float4*)(p);
    u.q[1] = *(const float4*)(p + 16);
    return u.v;
}

__device__ __forceinline__ v8f wmma_bf16(v16bf a, v16bf b, v8f c) {
    return __builtin_amdgcn_wmma_f32_16x16x32_bf16(
        false, a, false, b, (short)0, c, false, false);
}

// ---------------------------------------------------------------------------
// Prep kernels
// ---------------------------------------------------------------------------
__global__ __launch_bounds__(256) void cast_bf16_kernel(
    const float* __restrict__ x, __bf16* __restrict__ y) {
    long long i = ((long long)blockIdx.x * 256 + threadIdx.x) * 8;
    float4 a = *(const float4*)(x + i);
    float4 b = *(const float4*)(x + i + 4);
    union { __bf16 h[8]; float4 f; } u;
    u.h[0] = f2bf(a.x); u.h[1] = f2bf(a.y); u.h[2] = f2bf(a.z); u.h[3] = f2bf(a.w);
    u.h[4] = f2bf(b.x); u.h[5] = f2bf(b.y); u.h[6] = f2bf(b.z); u.h[7] = f2bf(b.w);
    *(float4*)(y + i) = u.f;
}

// Wt[n][k] = bf16(W[k][n])  (1024x1024)
__global__ __launch_bounds__(256) void transpose_cast_kernel(
    const float* __restrict__ W, __bf16* __restrict__ Wt) {
    int n = blockIdx.x * 16 + threadIdx.x;
    int k = blockIdx.y * 16 + threadIdx.y;
    Wt[(long long)n * DMODEL + k] = f2bf(W[(long long)k * DMODEL + n]);
}

// ---------------------------------------------------------------------------
// bf16 GEMM: C[8192x1024] = A[8192x1024] * BT[1024x1024]^T (+bias ...)
// Block tile 128x128, 8 waves (2 along M x 4 along N), K-step 32.
// Double-buffered async LDS staging.
// MODE 0: write Q/K layout [B,H,S,DEPTH] bf16 (+bias)
// MODE 1: write V^T layout [B,H,DEPTH,S] bf16 (+bias)
// MODE 2: write f32 [M,N] = acc + bias + residual
// ---------------------------------------------------------------------------
#define GEMM_LDT 40   // 32 + 8 pad (80B rows)

template <int MODE>
__global__ __launch_bounds__(256) void gemm_bf16_kernel(
    const __bf16* __restrict__ A, const __bf16* __restrict__ BT,
    const float* __restrict__ bias, const float* __restrict__ resid,
    void* __restrict__ outp) {
    __shared__ __align__(16) __bf16 As[2][128 * GEMM_LDT];
    __shared__ __align__(16) __bf16 Bs[2][128 * GEMM_LDT];

    const int tid  = threadIdx.x;
    const int lane = tid & 31;
    const int w    = tid >> 5;
    const int wm   = w & 1;                       // 0..1 -> 64 rows each
    const int wn   = w >> 1;                      // 0..3 -> 32 cols each
    const long long rowBase = (long long)blockIdx.y * 128;
    const int       colBase = blockIdx.x * 128;

    const unsigned long long aBase = (unsigned long long)A;
    const unsigned long long bBase = (unsigned long long)BT;

    v8f acc[4][2];
#pragma unroll
    for (int i = 0; i < 4; ++i)
#pragma unroll
        for (int j = 0; j < 2; ++j) acc[i][j] = vzero8();

    // Stage one K-slice (32 deep) of A and BT into buffer nb.
    auto stage = [&](int kt, int nb) {
#pragma unroll
        for (int i = 0; i < 2; ++i) {
            int cid = tid * 2 + i;                // 0..511
            int r   = cid >> 2;
            int c8  = (cid & 3) << 3;
            unsigned aoff = (unsigned)((((rowBase + r) * DMODEL) + kt * 32 + c8) * 2);
            unsigned boff = (unsigned)((((long long)(colBase + r) * DMODEL) + kt * 32 + c8) * 2);
            async_b128(lds_off(&As[nb][r * GEMM_LDT + c8]), aoff, aBase);
            async_b128(lds_off(&Bs[nb][r * GEMM_LDT + c8]), boff, bBase);
        }
    };

    stage(0, 0);
    for (int kt = 0; kt < DMODEL / 32; ++kt) {
        const int cur = kt & 1;
        wait_async0();            // this wave's copies into buffer `cur` done
        __syncthreads();          // everyone's copies done / prior reads done
        if (kt + 1 < DMODEL / 32) stage(kt + 1, cur ^ 1);  // overlap with WMMA

        const __bf16* as = As[cur];
        const __bf16* bs = Bs[cur];
        v16bf bfr[2];
#pragma unroll
        for (int jn = 0; jn < 2; ++jn)
            bfr[jn] = load_frag_b(&bs[(wn * 32 + jn * 16) * GEMM_LDT], GEMM_LDT, lane);
#pragma unroll
        for (int im = 0; im < 4; ++im) {
            v16bf afr = load_frag_a(&as[(wm * 64 + im * 16) * GEMM_LDT], GEMM_LDT, lane);
            acc[im][0] = wmma_bf16(afr, bfr[0], acc[im][0]);
            acc[im][1] = wmma_bf16(afr, bfr[1], acc[im][1]);
        }
    }

    const int hi = lane >> 4, ln = lane & 15;
#pragma unroll
    for (int im = 0; im < 4; ++im)
#pragma unroll
        for (int jn = 0; jn < 2; ++jn)
#pragma unroll
            for (int j = 0; j < 8; ++j) {
                long long r = rowBase + wm * 64 + im * 16 + j + hi * 8;
                int       c = colBase + wn * 32 + jn * 16 + ln;
                float v = acc[im][jn][j] + bias[c];
                if constexpr (MODE == 0) {
                    long long b = r >> 11, s = r & 2047;
                    int h = c >> 6, d = c & 63;
                    ((__bf16*)outp)[(((b * NHEADS + h) * SEQ + s) << 6) + d] = f2bf(v);
                } else if constexpr (MODE == 1) {
                    long long b = r >> 11, s = r & 2047;
                    int h = c >> 6, d = c & 63;
                    ((__bf16*)outp)[((b * NHEADS + h) * DEPTH + d) * SEQ + s] = f2bf(v);
                } else {
                    ((float*)outp)[r * DMODEL + c] = v + resid[r * DMODEL + c];
                }
            }
}

// ---------------------------------------------------------------------------
// Flash attention: one block = (b,h) x 128 query rows; 8 waves x 16 rows.
// Q frags live in registers; K tile / V^T tile staged in LDS via async copies;
// P tile round-trips through LDS (reusing K buffer) for C->A layout.
// ---------------------------------------------------------------------------
__global__ __launch_bounds__(256) void flash_attn_kernel(
    const __bf16* __restrict__ Qg, const __bf16* __restrict__ Kg,
    const __bf16* __restrict__ Vt, __bf16* __restrict__ O) {
    constexpr int LDK = 72;    // 64 + 8 pad  (K tile rows, 144B)
    constexpr int LDP = 136;   // 128 + 8 pad (P / V^T rows, 272B)
    __shared__ __align__(16) unsigned char smem[128 * LDP * 2 + 64 * LDP * 2];
    __bf16* KP = (__bf16*)smem;                        // K tile 128xLDK, then P 128xLDP
    __bf16* Vs = (__bf16*)(smem + 128 * LDP * 2);      // V^T tile 64xLDP

    const int tid = threadIdx.x, lane = tid & 31, w = tid >> 5;
    const int hi = lane >> 4, ln = lane & 15;
    const int bh = blockIdx.y;                         // b*16 + h
    const int qBase = blockIdx.x * 128;
    const long long skBase = (long long)bh * SEQ;

    const unsigned long long kgBase = (unsigned long long)Kg;
    const unsigned long long vtBase = (unsigned long long)Vt;

    // Q fragments straight from global (A-frag addressing pattern).
    v16bf qf[2];
    {
        const __bf16* qrow = Qg + (skBase + qBase + w * 16 + ln) * DEPTH;
#pragma unroll
        for (int kk = 0; kk < 2; ++kk) {
            const __bf16* p = qrow + kk * 32 + hi * 8;
            union { v16bf v; float4 q[2]; } u;
            u.q[0] = *(const float4*)(p);
            u.q[1] = *(const float4*)(p + 16);
            qf[kk] = u.v;
        }
    }

    v8f oacc[4];
    float mrow[8], lrow[8];
#pragma unroll
    for (int dn = 0; dn < 4; ++dn) oacc[dn] = vzero8();
#pragma unroll
    for (int j = 0; j < 8; ++j) { mrow[j] = -3.0e38f; lrow[j] = 0.0f; }

    for (int kt = 0; kt < SEQ / 128; ++kt) {
        const int kBase = kt * 128;
        // Async-stage K tile [128 keys][64] and V^T tile [64][128 keys].
#pragma unroll
        for (int i = 0; i < 4; ++i) {
            int cid = tid * 4 + i;                     // 0..1023
            int r = cid >> 3, c8 = (cid & 7) << 3;
            unsigned koff = (unsigned)((((skBase + kBase + r) * DEPTH) + c8) * 2);
            async_b128(lds_off(&KP[r * LDK + c8]), koff, kgBase);
        }
#pragma unroll
        for (int i = 0; i < 4; ++i) {
            int cid = tid * 4 + i;
            int r = cid >> 4, c8 = (cid & 15) << 3;
            unsigned voff = (unsigned)(((((long long)bh * DEPTH + r) * SEQ) + kBase + c8) * 2);
            async_b128(lds_off(&Vs[r * LDP + c8]), voff, vtBase);
        }
        wait_async0();
        __syncthreads();

        // S = Q * K^T   (8 col-tiles x 2 k-chunks = 16 WMMA / wave)
        v8f s[8];
#pragma unroll
        for (int n = 0; n < 8; ++n) s[n] = vzero8();
#pragma unroll
        for (int kk = 0; kk < 2; ++kk)
#pragma unroll
            for (int n = 0; n < 8; ++n) {
                v16bf kf = load_frag_b(&KP[(n * 16) * LDK + kk * 32], LDK, lane);
                s[n] = wmma_bf16(qf[kk], kf, s[n]);
            }

        // Online softmax (row stats per C-layout row j + 8*hi, reduced across
        // the 16 lanes holding that row).
        constexpr float scale = 0.125f;                // 1/sqrt(64)
#pragma unroll
        for (int j = 0; j < 8; ++j) {
            float pm = -3.0e38f;
#pragma unroll
            for (int n = 0; n < 8; ++n) { s[n][j] *= scale; pm = fmaxf(pm, s[n][j]); }
#pragma unroll
            for (int msk = 1; msk < 16; msk <<= 1)
                pm = fmaxf(pm, __shfl_xor(pm, msk, 32));
            float nm   = fmaxf(mrow[j], pm);
            float corr = __expf(mrow[j] - nm);
            mrow[j] = nm;
            float psum = 0.0f;
#pragma unroll
            for (int n = 0; n < 8; ++n) {
                float p = __expf(s[n][j] - nm);
                s[n][j] = p; psum += p;
            }
#pragma unroll
            for (int msk = 1; msk < 16; msk <<= 1)
                psum += __shfl_xor(psum, msk, 32);
            lrow[j] = lrow[j] * corr + psum;
#pragma unroll
            for (int dn = 0; dn < 4; ++dn) oacc[dn][j] *= corr;
        }
        __syncthreads();   // all waves done reading K before P overwrites it

        // P (bf16) -> LDS, C-layout scatter.
#pragma unroll
        for (int n = 0; n < 8; ++n)
#pragma unroll
            for (int j = 0; j < 8; ++j)
                KP[(w * 16 + j + hi * 8) * LDP + n * 16 + ln] = f2bf(s[n][j]);
        __syncthreads();

        // O += P * V   (4 k-chunks x 4 d-tiles = 16 WMMA / wave)
#pragma unroll
        for (int kc = 0; kc < 4; ++kc) {
            v16bf pf = load_frag_a(&KP[(w * 16) * LDP + kc * 32], LDP, lane);
#pragma unroll
            for (int dn = 0; dn < 4; ++dn) {
                v16bf vf = load_frag_b(&Vs[(dn * 16) * LDP + kc * 32], LDP, lane);
                oacc[dn] = wmma_bf16(pf, vf, oacc[dn]);
            }
        }
        __syncthreads();   // done reading P/Vs before next async stage
    }

    // Normalize and merge heads: attnOut[b*S + q][h*64 + d] bf16.
    const int b = bh >> 4, h = bh & 15;
#pragma unroll
    for (int dn = 0; dn < 4; ++dn)
#pragma unroll
        for (int j = 0; j < 8; ++j) {
            float inv = 1.0f / lrow[j];
            long long g = (long long)b * SEQ + qBase + w * 16 + j + hi * 8;
            int c = h * DEPTH + dn * 16 + ln;
            O[g * DMODEL + c] = f2bf(oacc[dn][j] * inv);
        }
}

// ---------------------------------------------------------------------------
// LayerNorm over rows of 1024: out = (y - mu) * rsqrt(var+eps) * gamma + beta
// ---------------------------------------------------------------------------
__global__ __launch_bounds__(256) void layernorm_kernel(
    const float* __restrict__ Y, const float* __restrict__ gamma,
    const float* __restrict__ beta, float* __restrict__ out) {
    __shared__ float sbuf[256];
    const long long row = blockIdx.x;
    const float* y = Y + row * DMODEL;
    const int tid = threadIdx.x;

    float v[4]; float s = 0.0f;
#pragma unroll
    for (int i = 0; i < 4; ++i) { v[i] = y[tid + 256 * i]; s += v[i]; }
    sbuf[tid] = s; __syncthreads();
    for (int off = 128; off > 0; off >>= 1) {
        if (tid < off) sbuf[tid] += sbuf[tid + off];
        __syncthreads();
    }
    float mu = sbuf[0] * (1.0f / DMODEL);
    __syncthreads();

    float q = 0.0f;
#pragma unroll
    for (int i = 0; i < 4; ++i) { float d = v[i] - mu; q += d * d; }
    sbuf[tid] = q; __syncthreads();
    for (int off = 128; off > 0; off >>= 1) {
        if (tid < off) sbuf[tid] += sbuf[tid + off];
        __syncthreads();
    }
    float rstd = rsqrtf(sbuf[0] * (1.0f / DMODEL) + 1e-6f);
#pragma unroll
    for (int i = 0; i < 4; ++i) {
        int c = tid + 256 * i;
        out[row * DMODEL + c] = (v[i] - mu) * rstd * gamma[c] + beta[c];
    }
}

// ---------------------------------------------------------------------------
// Host launch
// ---------------------------------------------------------------------------
extern "C" void kernel_launch(void* const* d_in, const int* in_sizes, int n_in,
                              void* d_out, int out_size, void* d_ws, size_t ws_size,
                              hipStream_t stream) {
    const float* X     = (const float*)d_in[0];
    const float* Wq    = (const float*)d_in[1];
    const float* bq    = (const float*)d_in[2];
    const float* Wk    = (const float*)d_in[3];
    const float* bk    = (const float*)d_in[4];
    const float* Wv    = (const float*)d_in[5];
    const float* bv    = (const float*)d_in[6];
    const float* Wo    = (const float*)d_in[7];
    const float* bo    = (const float*)d_in[8];
    const float* gamma = (const float*)d_in[9];
    const float* beta  = (const float*)d_in[10];
    float* outF = (float*)d_out;

    const size_t MB = (size_t)1 << 20;
    char* ws = (char*)d_ws;
    __bf16* Xb   = (__bf16*)(ws + 0);          // 16 MB
    __bf16* WqT  = (__bf16*)(ws + 16 * MB);    //  2 MB
    __bf16* WkT  = (__bf16*)(ws + 18 * MB);
    __bf16* WvT  = (__bf16*)(ws + 20 * MB);
    __bf16* WoT  = (__bf16*)(ws + 22 * MB);
    __bf16* Qb   = (__bf16*)(ws + 24 * MB);    // 16 MB
    __bf16* Kb   = (__bf16*)(ws + 40 * MB);    // 16 MB
    __bf16* Vtb  = (__bf16*)(ws + 56 * MB);    // 16 MB
    __bf16* AOb  = (__bf16*)(ws + 72 * MB);    // 16 MB
    float*  Yf   = (float*)(ws + 88 * MB);     // 32 MB

    // 1) casts / weight transposes
    cast_bf16_kernel<<<dim3((MROWS * DMODEL) / (256 * 8)), dim3(256), 0, stream>>>(X, Xb);
    dim3 tb(16, 16), tg(DMODEL / 16, DMODEL / 16);
    transpose_cast_kernel<<<tg, tb, 0, stream>>>(Wq, WqT);
    transpose_cast_kernel<<<tg, tb, 0, stream>>>(Wk, WkT);
    transpose_cast_kernel<<<tg, tb, 0, stream>>>(Wv, WvT);
    transpose_cast_kernel<<<tg, tb, 0, stream>>>(Wo, WoT);

    // 2) QKV projections
    dim3 gg(DMODEL / 128, MROWS / 128);
    gemm_bf16_kernel<0><<<gg, dim3(256), 0, stream>>>(Xb, WqT, bq, nullptr, (void*)Qb);
    gemm_bf16_kernel<0><<<gg, dim3(256), 0, stream>>>(Xb, WkT, bk, nullptr, (void*)Kb);
    gemm_bf16_kernel<1><<<gg, dim3(256), 0, stream>>>(Xb, WvT, bv, nullptr, (void*)Vtb);

    // 3) flash attention
    flash_attn_kernel<<<dim3(SEQ / 128, BATCH * NHEADS), dim3(256), 0, stream>>>(
        Qb, Kb, Vtb, AOb);

    // 4) output projection + bias + residual (f32)
    gemm_bf16_kernel<2><<<gg, dim3(256), 0, stream>>>(AOb, WoT, bo, X, (void*)Yf);

    // 5) layernorm -> d_out
    layernorm_kernel<<<dim3(MROWS), dim3(256), 0, stream>>>(Yf, gamma, beta, outF);
}
